// SCPredictor_61194694033417
// MI455X (gfx1250) — compile-verified, matrix operand order checked by reading, not run on hardware
//
#include <hip/hip_runtime.h>

typedef __attribute__((ext_vector_type(16))) __bf16 v16bf;
typedef __attribute__((ext_vector_type(8)))  __bf16 v8bf;
typedef __attribute__((ext_vector_type(8)))  float  v8f;
typedef __attribute__((ext_vector_type(4)))  unsigned int su4;
typedef __attribute__((ext_vector_type(8)))  unsigned int su8;

union FragBF { v16bf v; v8bf h[2]; };

#define BB 32
#define NN 1024
#define DD 512
#define NCLS 4

// ---------------------------------------------------------------------------
// 1) column degrees -> dis = rsqrt(deg)
// ---------------------------------------------------------------------------
__global__ __launch_bounds__(256) void k_deg(const float* __restrict__ A,
                                             float* __restrict__ dis) {
  int idx = blockIdx.x * 256 + threadIdx.x;        // b*N + j
  int b = idx >> 10, j = idx & (NN - 1);
  const float* p = A + (size_t)b * NN * NN + j;
  float s = 0.f;
  for (int i = 0; i < NN; ++i) s += p[(size_t)i * NN];
  dis[idx] = (s > 0.f) ? rsqrtf(s) : 0.f;
}

// ---------------------------------------------------------------------------
// 2) X = bf16(A); An = bf16(A^T * dis_i * dis_j)   (LDS tile transpose)
// ---------------------------------------------------------------------------
__global__ __launch_bounds__(256) void k_prep(const float* __restrict__ A,
                                              const float* __restrict__ dis,
                                              __bf16* __restrict__ X,
                                              __bf16* __restrict__ An) {
  __shared__ float tile[32][33];
  int tx = threadIdx.x, ty = threadIdx.y;
  int i0 = blockIdx.x * 32, j0 = blockIdx.y * 32, b = blockIdx.z;
  const float* Ab = A + (size_t)b * NN * NN;
  __bf16* Xb = X + (size_t)b * NN * NN;
  __bf16* Anb = An + (size_t)b * NN * NN;
  for (int r = ty; r < 32; r += 8) {
    float v = Ab[(size_t)(j0 + r) * NN + i0 + tx];
    tile[r][tx] = v;
    Xb[(size_t)(j0 + r) * NN + i0 + tx] = (__bf16)v;
  }
  __syncthreads();
  for (int r = ty; r < 32; r += 8) {
    float di = dis[b * NN + i0 + r];
    float dj = dis[b * NN + j0 + tx];
    Anb[(size_t)(i0 + r) * NN + j0 + tx] = (__bf16)(tile[tx][r] * di * dj);
  }
}

// ---------------------------------------------------------------------------
// 3) weight convert + transpose: W[R,C] -> Wt[C,R] bf16
// ---------------------------------------------------------------------------
__global__ __launch_bounds__(256) void k_cvt_t(const float* __restrict__ W,
                                               __bf16* __restrict__ Wt,
                                               int R, int C) {
  int idx = blockIdx.x * 256 + threadIdx.x;
  int r = idx % R, c = idx / R;
  Wt[idx] = (__bf16)W[(size_t)r * C + c];
}

// ---------------------------------------------------------------------------
// TDM: DMA one 128-row x 32-col bf16 tile (row stride K elems) into LDS,
// padding each 64B row with 16B so the LDS row stride is LDT=40 elements.
// D# per CDNA5 ISA ch.8: group0 {count, lds_addr, global_addr, type=2},
// group1 {data_size=2B, pad_enable, pad_interval=16DW, pad_amount=4DW,
//         tensor_dim0=K, tensor_dim1=128, tile_dim0=32, tile_dim1=128,
//         tensor_dim0_stride=K}.  Issued per-wave (EXEC ignored).
// ---------------------------------------------------------------------------
__device__ __forceinline__ void tdm_load_tile(const __bf16* gsrc,
                                              unsigned ldsAddr, int K) {
  unsigned long long ga = (unsigned long long)(uintptr_t)gsrc;
  su4 g0;
  g0[0] = 1u;                                   // count=1 (valid user D#)
  g0[1] = ldsAddr;                              // lds_addr (bytes)
  g0[2] = (unsigned)ga;                         // global_addr[31:0]
  g0[3] = (unsigned)(ga >> 32) | (2u << 30);    // global_addr[56:32] | type=2
  su8 g1;
  g1[0] = (1u << 16)    // data_size = 2 bytes
        | (1u << 20)    // pad_enable
        | (3u << 22)    // pad_interval: 16 DWORDs (64B)
        | (3u << 25);   // pad_amount:    4 DWORDs (16B)
  g1[1] = ((unsigned)K & 0xffffu) << 16;                  // tensor_dim0 lo16
  g1[2] = ((unsigned)K >> 16) | (128u << 16);             // dim0 hi | dim1 lo16
  g1[3] = (32u << 16);                                    // dim1 hi=0 | tile_dim0=32
  g1[4] = 128u;                                           // tile_dim1=128, tile_dim2=0
  g1[5] = (unsigned)K;                                    // tensor_dim0_stride lo32
  g1[6] = 0u;                                             // stride hi | dim1_stride lo
  g1[7] = 0u;
  asm volatile("tensor_load_to_lds %0, %1" :: "s"(g0), "s"(g1) : "memory");
}

// ---------------------------------------------------------------------------
// 4) TN bf16 WMMA GEMM: C[m,n] = sum_k A[m,k] * Bt[n,k]
//    Tiles DMA'd by the Tensor Data Mover, double-buffered on TENSORcnt.
//    MODE 0: store transposed bf16 Ct[n*M+m]       (no bias)
//    MODE 1: store bf16 Out[m*Nc+n], bias + relu
//    MODE 2: store f32  Out[m*Nc+n], bias
// ---------------------------------------------------------------------------
#define BK 32
#define LDT 40

template <int MODE>
__global__ __launch_bounds__(256) void k_gemm(
    const __bf16* __restrict__ A, long long aStrideB,
    const __bf16* __restrict__ Bt, long long bStrideB,
    void* __restrict__ OutV, long long oStrideB,
    const float* __restrict__ bias, int M, int Nc, int K) {
  __shared__ __align__(16) __bf16 As[2][128 * LDT];
  __shared__ __align__(16) __bf16 Bs[2][128 * LDT];

  const int tid = threadIdx.x;
  const int b = blockIdx.z;
  const int n0 = blockIdx.x * 128;
  const int m0 = blockIdx.y * 128;
  const int wave = tid >> 5, lane = tid & 31;   // wave32
  const int wm = wave & 1, wn = wave >> 1;      // 2 x 4 wave grid
  const int hf = lane >> 4, lrow = lane & 15;

  const __bf16* Ab = A + (size_t)b * aStrideB + (size_t)m0 * K;
  const __bf16* Bb = Bt + (size_t)b * bStrideB + (size_t)n0 * K;

  const unsigned ldsA0 = (unsigned)(uintptr_t)&As[0][0];
  const unsigned ldsA1 = (unsigned)(uintptr_t)&As[1][0];
  const unsigned ldsB0 = (unsigned)(uintptr_t)&Bs[0][0];
  const unsigned ldsB1 = (unsigned)(uintptr_t)&Bs[1][0];

  v8f acc[4][2] = {};
  const int nk = K / BK;

  if (tid < 32) {                       // wave 0 drives the TDM
    tdm_load_tile(Ab, ldsA0, K);
    tdm_load_tile(Bb, ldsB0, K);
  }

  for (int kt = 0; kt < nk; ++kt) {
    const int cur = kt & 1;
    if (tid < 32) {
      if (kt + 1 < nk) {                // prefetch next K-slab via TDM
        tdm_load_tile(Ab + (size_t)(kt + 1) * BK, cur ? ldsA0 : ldsA1, K);
        tdm_load_tile(Bb + (size_t)(kt + 1) * BK, cur ? ldsB0 : ldsB1, K);
        __builtin_amdgcn_s_wait_tensorcnt(2);   // tile kt landed (in-order)
      } else {
        __builtin_amdgcn_s_wait_tensorcnt(0);
      }
    }
    __syncthreads();

    const __bf16* Asb = &As[cur][0];
    const __bf16* Bsb = &Bs[cur][0];
    FragBF afr[4], bfr[2];
#pragma unroll
    for (int sm = 0; sm < 4; ++sm) {
      int base = (wm * 64 + sm * 16 + lrow) * LDT + hf * 8;
      afr[sm].h[0] = *(const v8bf*)&Asb[base];
      afr[sm].h[1] = *(const v8bf*)&Asb[base + 16];
    }
#pragma unroll
    for (int sn = 0; sn < 2; ++sn) {
      int base = (wn * 32 + sn * 16 + lrow) * LDT + hf * 8;
      bfr[sn].h[0] = *(const v8bf*)&Bsb[base];
      bfr[sn].h[1] = *(const v8bf*)&Bsb[base + 16];
    }
#pragma unroll
    for (int sm = 0; sm < 4; ++sm)
#pragma unroll
      for (int sn = 0; sn < 2; ++sn)
        acc[sm][sn] = __builtin_amdgcn_wmma_f32_16x16x32_bf16(
            false, afr[sm].v, false, bfr[sn].v, (short)0, acc[sm][sn],
            false, false);
    __syncthreads();    // protect buffer about to be overwritten by TDM
  }

#pragma unroll
  for (int sm = 0; sm < 4; ++sm) {
#pragma unroll
    for (int sn = 0; sn < 2; ++sn) {
      int n = n0 + wn * 32 + sn * 16 + lrow;
      int mb = m0 + wm * 64 + sm * 16 + hf * 8;   // 8 consecutive m per lane
      if (MODE == 0) {
        __bf16* Ct = (__bf16*)OutV + (size_t)b * oStrideB;
        v8bf pk;
#pragma unroll
        for (int r = 0; r < 8; ++r) pk[r] = (__bf16)acc[sm][sn][r];
        *(v8bf*)&Ct[(size_t)n * M + mb] = pk;
      } else if (MODE == 1) {
        __bf16* Outp = (__bf16*)OutV + (size_t)b * oStrideB;
        float bn = bias[n];
#pragma unroll
        for (int r = 0; r < 8; ++r) {
          float v = fmaxf(acc[sm][sn][r] + bn, 0.f);
          Outp[(size_t)(mb + r) * Nc + n] = (__bf16)v;
        }
      } else {
        float* Outp = (float*)OutV + (size_t)b * oStrideB;
        float bn = bias[n];
#pragma unroll
        for (int r = 0; r < 8; ++r)
          Outp[(size_t)(mb + r) * Nc + n] = acc[sm][sn][r] + bn;
      }
    }
  }
}

// ---------------------------------------------------------------------------
// 5) in-place LayerNorm over D=512 per node row
// ---------------------------------------------------------------------------
__global__ __launch_bounds__(256) void k_lnrow(float* __restrict__ H,
                                               const float* __restrict__ g,
                                               const float* __restrict__ tt) {
  __shared__ float s1[256], s2[256];
  float* h = H + (size_t)blockIdx.x * DD;
  int tid = threadIdx.x;
  float x0 = h[tid], x1 = h[tid + 256];
  s1[tid] = x0 + x1;
  s2[tid] = x0 * x0 + x1 * x1;
  __syncthreads();
  for (int o = 128; o > 0; o >>= 1) {
    if (tid < o) { s1[tid] += s1[tid + o]; s2[tid] += s2[tid + o]; }
    __syncthreads();
  }
  float mu = s1[0] * (1.f / 512.f);
  float var = s2[0] * (1.f / 512.f) - mu * mu;
  float rs = rsqrtf(var + 1e-5f);
  h[tid] = (x0 - mu) * rs * g[tid] + tt[tid];
  h[tid + 256] = (x1 - mu) * rs * g[tid + 256] + tt[tid + 256];
}

// ---------------------------------------------------------------------------
// 6) mean-pool over N nodes -> z_pooled [B,D] (ws copy + d_out copy)
// ---------------------------------------------------------------------------
__global__ __launch_bounds__(256) void k_pool(const float* __restrict__ Z,
                                              float* __restrict__ zpool,
                                              float* __restrict__ outz) {
  int idx = blockIdx.x * 256 + threadIdx.x;  // b*D + d
  int b = idx / DD, d = idx % DD;
  const float* p = Z + (size_t)b * NN * DD + d;
  float s = 0.f;
  for (int n = 0; n < NN; ++n) s += p[(size_t)n * DD];
  s *= (1.f / (float)NN);
  zpool[idx] = s;
  outz[idx] = s;
}

// ---------------------------------------------------------------------------
// 7) tiny classifier MLP (one block per batch row)
// ---------------------------------------------------------------------------
__global__ __launch_bounds__(128) void k_cls(
    const float* __restrict__ zpool, const float* __restrict__ Wc1,
    const float* __restrict__ bc1, const float* __restrict__ g1,
    const float* __restrict__ t1, const float* __restrict__ Wc2,
    const float* __restrict__ bc2, const float* __restrict__ g2,
    const float* __restrict__ t2, const float* __restrict__ Wc3,
    const float* __restrict__ bc3, float* __restrict__ logits) {
  __shared__ float zr[512], c1s[128], c2s[64], r1[128], r2[128];
  int b = blockIdx.x, tid = threadIdx.x;
  for (int d = tid; d < 512; d += 128) zr[d] = zpool[b * 512 + d];
  __syncthreads();

  float a = bc1[tid];
  for (int k = 0; k < 512; ++k) a += zr[k] * Wc1[k * 128 + tid];
  r1[tid] = a; r2[tid] = a * a;
  __syncthreads();
  for (int o = 64; o > 0; o >>= 1) {
    if (tid < o) { r1[tid] += r1[tid + o]; r2[tid] += r2[tid + o]; }
    __syncthreads();
  }
  float mu = r1[0] * (1.f / 128.f);
  float var = r2[0] * (1.f / 128.f) - mu * mu;
  float rs = rsqrtf(var + 1e-5f);
  __syncthreads();
  c1s[tid] = fmaxf(0.f, (a - mu) * rs * g1[tid] + t1[tid]);
  __syncthreads();

  float a2 = 0.f;
  if (tid < 64) {
    a2 = bc2[tid];
    for (int k = 0; k < 128; ++k) a2 += c1s[k] * Wc2[k * 64 + tid];
  }
  r1[tid] = (tid < 64) ? a2 : 0.f;
  r2[tid] = (tid < 64) ? a2 * a2 : 0.f;
  __syncthreads();
  for (int o = 64; o > 0; o >>= 1) {
    if (tid < o) { r1[tid] += r1[tid + o]; r2[tid] += r2[tid + o]; }
    __syncthreads();
  }
  float mu2 = r1[0] * (1.f / 64.f);
  float var2 = r2[0] * (1.f / 64.f) - mu2 * mu2;
  float rs2 = rsqrtf(var2 + 1e-5f);
  __syncthreads();
  if (tid < 64) c2s[tid] = fmaxf(0.f, (a2 - mu2) * rs2 * g2[tid] + t2[tid]);
  __syncthreads();
  if (tid < NCLS) {
    float l = bc3[tid];
    for (int k = 0; k < 64; ++k) l += c2s[k] * Wc3[k * 4 + tid];
    logits[b * NCLS + tid] = l;
  }
}

// ---------------------------------------------------------------------------
extern "C" void kernel_launch(void* const* d_in, const int* in_sizes, int n_in,
                              void* d_out, int out_size, void* d_ws,
                              size_t ws_size, hipStream_t stream) {
  (void)in_sizes; (void)n_in; (void)out_size; (void)ws_size;
  const float* A    = (const float*)d_in[0];
  const float* W1   = (const float*)d_in[1];
  const float* b1   = (const float*)d_in[2];
  const float* W2   = (const float*)d_in[3];
  const float* b2   = (const float*)d_in[4];
  const float* ln_g = (const float*)d_in[5];
  const float* ln_b = (const float*)d_in[6];
  const float* Wc1  = (const float*)d_in[7];
  const float* bc1  = (const float*)d_in[8];
  const float* g1   = (const float*)d_in[9];
  const float* t1   = (const float*)d_in[10];
  const float* Wc2  = (const float*)d_in[11];
  const float* bc2  = (const float*)d_in[12];
  const float* g2   = (const float*)d_in[13];
  const float* t2   = (const float*)d_in[14];
  const float* Wc3  = (const float*)d_in[15];
  const float* bc3  = (const float*)d_in[16];
  float* out = (float*)d_out;

  char* w = (char*)d_ws;
  auto alloc = [&](size_t bytes) {
    void* p = (void*)w;
    w += (bytes + 255) & ~(size_t)255;
    return p;
  };
  float*  dis  = (float*)alloc((size_t)BB * NN * 4);
  __bf16* X    = (__bf16*)alloc((size_t)BB * NN * NN * 2);
  __bf16* An   = (__bf16*)alloc((size_t)BB * NN * NN * 2);
  __bf16* W1t  = (__bf16*)alloc((size_t)NN * DD * 2);
  __bf16* W2t  = (__bf16*)alloc((size_t)DD * DD * 2);
  __bf16* C1t  = (__bf16*)alloc((size_t)BB * DD * NN * 2);
  __bf16* H1   = (__bf16*)alloc((size_t)BB * NN * DD * 2);
  __bf16* C2t  = (__bf16*)alloc((size_t)BB * DD * NN * 2);
  float*  H2   = (float*)alloc((size_t)BB * NN * DD * 4);
  float*  zpl  = (float*)alloc((size_t)BB * DD * 4);

  // 1) degrees
  k_deg<<<(BB * NN) / 256, 256, 0, stream>>>(A, dis);
  // 2) X bf16 + normalized-transposed An bf16
  k_prep<<<dim3(NN / 32, NN / 32, BB), dim3(32, 8), 0, stream>>>(A, dis, X, An);
  // 3) weights -> transposed bf16
  k_cvt_t<<<(NN * DD) / 256, 256, 0, stream>>>(W1, W1t, NN, DD);
  k_cvt_t<<<(DD * DD) / 256, 256, 0, stream>>>(W2, W2t, DD, DD);

  dim3 gg(DD / 128, NN / 128, BB);
  // 4) C1t = (X @ W1)^T                 [B, D, N] bf16
  k_gemm<0><<<gg, 256, 0, stream>>>(X, (long long)NN * NN, W1t, 0, C1t,
                                    (long long)DD * NN, nullptr, NN, DD, NN);
  // 5) H1 = relu(An @ C1 + b1)          [B, N, D] bf16
  k_gemm<1><<<gg, 256, 0, stream>>>(An, (long long)NN * NN, C1t,
                                    (long long)DD * NN, H1, (long long)NN * DD,
                                    b1, NN, DD, NN);
  // 6) C2t = (H1 @ W2)^T                [B, D, N] bf16
  k_gemm<0><<<gg, 256, 0, stream>>>(H1, (long long)NN * DD, W2t, 0, C2t,
                                    (long long)DD * NN, nullptr, NN, DD, DD);
  // 7) H2 = An @ C2 + b2                [B, N, D] f32
  k_gemm<2><<<gg, 256, 0, stream>>>(An, (long long)NN * NN, C2t,
                                    (long long)DD * NN, H2, (long long)NN * DD,
                                    b2, NN, DD, NN);
  // 8) layernorm each node row (in place)
  k_lnrow<<<BB * NN, 256, 0, stream>>>(H2, ln_g, ln_b);
  // 9) mean pool -> zpool (ws) and d_out[128..]
  k_pool<<<(BB * DD) / 256, 256, 0, stream>>>(H2, zpl, out + BB * NCLS);
  // 10) classifier MLP -> logits d_out[0..127]
  k_cls<<<BB, 128, 0, stream>>>(zpl, Wc1, bc1, g1, t1, Wc2, bc2, g2, t2, Wc3,
                                bc3, out);
}
